// SelfAttention_72206990180782
// MI455X (gfx1250) — matrix operand order repaired and checked by reading, not compiled, not run on hardware
//
#include <hip/hip_runtime.h>
#include <hip/hip_bf16.h>
#include <math.h>

typedef __attribute__((ext_vector_type(16))) _Float16 v16h;
typedef __attribute__((ext_vector_type(8)))  _Float16 v8h;
typedef __attribute__((ext_vector_type(8)))  float    v8f;

#define HDIM   768
#define SEQ    512
#define BATCH  8
#define NH     12
#define DH     64
#define MTOT   (BATCH*SEQ)          /* 4096 */

/* workspace layout, in _Float16 elements */
#define XH_OFF   0
#define WT_OFF   (MTOT*HDIM)                /* 3,145,728  */
#define QKV_OFF  (WT_OFF + 3*HDIM*HDIM)     /* +1,769,472 */

__device__ __forceinline__ v8f wmma16(v16h a, v16h b, v8f c) {
  return __builtin_amdgcn_wmma_f32_16x16x32_f16(
      /*neg_a=*/false, a, /*neg_b=*/false, b,
      /*c_mod=*/(short)0, c, /*reuse_a=*/false, /*reuse_b=*/false);
}

/* A-matrix 16x32 f16 fragment: lane holds row (lane&15); element i<8 -> K=kb+g*8+i,
   element i>=8 -> K=kb+16+g*8+(i-8)  (ISA 05_wmma 16-bit A layout) */
__device__ __forceinline__ v16h load_a(const _Float16* __restrict__ row, int kb, int g) {
  v8h c0 = *(const v8h*)(row + kb + g * 8);
  v8h c1 = *(const v8h*)(row + kb + 16 + g * 8);
  v16h r;
#pragma unroll
  for (int i = 0; i < 8; ++i) { r[i] = c0[i]; r[i + 8] = c1[i]; }
  return r;
}

/* B-matrix 32x16 f16 fragment: lane holds column (lane&15); element i<8 -> K=kb+g*8+i,
   element i>=8 -> K=kb+16+g*8+(i-8) -> two b128 loads (operand stored K-contiguous per column) */
__device__ __forceinline__ v16h load_b(const _Float16* __restrict__ col, int kb, int g) {
  v8h c0 = *(const v8h*)(col + kb + g * 8);
  v8h c1 = *(const v8h*)(col + kb + 16 + g * 8);
  v16h r;
#pragma unroll
  for (int i = 0; i < 8; ++i) { r[i] = c0[i]; r[i + 8] = c1[i]; }
  return r;
}

/* ---- fp32 -> f16 convert of hidden_states ---- */
__global__ void cvt_x_kernel(const float* __restrict__ x, _Float16* __restrict__ xh, int n) {
  int i = blockIdx.x * blockDim.x + threadIdx.x;
  if (i < n) xh[i] = (_Float16)x[i];
}

/* ---- W[in][out] fp32 -> Wt[out][in] f16, tiled transpose via LDS ---- */
__global__ void wt_kernel(const float* __restrict__ w0, const float* __restrict__ w1,
                          const float* __restrict__ w2, _Float16* __restrict__ wt) {
  __shared__ float tile[32][33];
  const float* W = (blockIdx.z == 0) ? w0 : (blockIdx.z == 1 ? w1 : w2);
  _Float16* Wt = wt + (size_t)blockIdx.z * (HDIM * HDIM);
  int n0 = blockIdx.x * 32, k0 = blockIdx.y * 32;
  int tx = threadIdx.x, ty = threadIdx.y;        /* block = (32, 8) */
#pragma unroll
  for (int j = 0; j < 32; j += 8)
    tile[ty + j][tx] = W[(size_t)(k0 + ty + j) * HDIM + (n0 + tx)];
  __syncthreads();
#pragma unroll
  for (int j = 0; j < 32; j += 8)
    Wt[(size_t)(n0 + ty + j) * HDIM + (k0 + tx)] = (_Float16)tile[tx][ty + j];
}

/* ---- projection GEMM: Q/K/V[m,n] = X[m,:] . W[:,n] + bias[n], f16 WMMA, f32 acc ----
   grid (6, 32, 3), block 256: 128x128 tile, 8 waves of 32x64 (2 m-tiles x 4 n-tiles) */
__global__ __launch_bounds__(256) void proj_kernel(
    const _Float16* __restrict__ xh, const _Float16* __restrict__ wt,
    const float* __restrict__ b0, const float* __restrict__ b1, const float* __restrict__ b2,
    _Float16* __restrict__ qkv) {
  int p = blockIdx.z;
  const _Float16* Wt = wt + (size_t)p * (HDIM * HDIM);
  const float* bias = (p == 0) ? b0 : (p == 1 ? b1 : b2);
  _Float16* O = qkv + (size_t)p * (MTOT * HDIM);

  int tid = threadIdx.x, w = tid >> 5, lane = tid & 31, g = lane >> 4, l15 = lane & 15;
  int wm = w & 3, wn = w >> 2;
  int row0 = blockIdx.y * 128 + wm * 32;
  int col0 = blockIdx.x * 128 + wn * 64;

  v8f acc[2][4] = {};
  const _Float16* arow0 = xh + (size_t)(row0 + l15) * HDIM;
  const _Float16* arow1 = xh + (size_t)(row0 + 16 + l15) * HDIM;
  const _Float16* brow[4];
#pragma unroll
  for (int nt = 0; nt < 4; ++nt)
    brow[nt] = Wt + (size_t)(col0 + nt * 16 + l15) * HDIM;

  for (int kb = 0; kb < HDIM; kb += 32) {
    v16h a0 = load_a(arow0, kb, g);
    v16h a1 = load_a(arow1, kb, g);
#pragma unroll
    for (int nt = 0; nt < 4; ++nt) {
      v16h bb = load_b(brow[nt], kb, g);
      acc[0][nt] = wmma16(a0, bb, acc[0][nt]);
      acc[1][nt] = wmma16(a1, bb, acc[1][nt]);
    }
  }

#pragma unroll
  for (int mt = 0; mt < 2; ++mt)
#pragma unroll
    for (int nt = 0; nt < 4; ++nt) {
      int n = col0 + nt * 16 + l15;
      float bv = bias[n];
#pragma unroll
      for (int r = 0; r < 8; ++r) {
        int m = row0 + mt * 16 + r + 8 * g;   /* D layout: lanes16-31 hold M=r+8 */
        O[(size_t)m * HDIM + n] = (_Float16)(acc[mt][nt][r] + bv);
      }
    }
}

/* ---- fused Gram-matrix + mask + softmax ----
   grid (32 strips, 12 heads, 3*8), block 256; strip = 16 query rows x 512 keys.
   Each wave owns a 16x64 slab (4 n-tiles), K-loop over dH=64 in two 32-steps. */
__global__ __launch_bounds__(256) void attn_kernel(
    const _Float16* __restrict__ qkv, const float* __restrict__ mask,
    float* __restrict__ out) {
  __shared__ float sc[16 * 516];                /* 33 KB, padded stride vs bank conflicts */
  int strip = blockIdx.x;
  int h = blockIdx.y;
  int p = blockIdx.z >> 3;
  int b = blockIdx.z & 7;
  const _Float16* Qh = qkv + (size_t)p * (MTOT * HDIM) + (size_t)b * SEQ * HDIM + h * DH;

  int tid = threadIdx.x, w = tid >> 5, lane = tid & 31, g = lane >> 4, l15 = lane & 15;
  int col0 = w * 64;

  v8f acc[4] = {};
  const _Float16* arow = Qh + (size_t)(strip * 16 + l15) * HDIM;
  const _Float16* brow[4];
#pragma unroll
  for (int nt = 0; nt < 4; ++nt)
    brow[nt] = Qh + (size_t)(col0 + nt * 16 + l15) * HDIM;   /* symmetric operand: B = Q */

#pragma unroll
  for (int kb = 0; kb < DH; kb += 32) {
    v16h a = load_a(arow, kb, g);
#pragma unroll
    for (int nt = 0; nt < 4; ++nt)
      acc[nt] = wmma16(a, load_b(brow[nt], kb, g), acc[nt]);
  }

#pragma unroll
  for (int nt = 0; nt < 4; ++nt)
#pragma unroll
    for (int r = 0; r < 8; ++r)
      sc[(r + 8 * g) * 516 + col0 + nt * 16 + l15] = acc[nt][r];
  __syncthreads();

  /* softmax: wave w handles strip rows w*2 and w*2+1, 16 cols per lane */
  const float scale = 0.125f;                   /* 1/sqrt(64) */
  size_t obase = ((size_t)(p * BATCH + b) * NH + h) * ((size_t)SEQ * SEQ);
#pragma unroll
  for (int rr = 0; rr < 2; ++rr) {
    int row = w * 2 + rr;
    float vv[16];
    float mx = -3.0e38f;
#pragma unroll
    for (int i = 0; i < 16; ++i) {
      int c = lane + i * 32;
      float s = sc[row * 516 + c] * scale + (1.0f - mask[b * SEQ + c]) * -10000.0f;
      vv[i] = s;
      mx = fmaxf(mx, s);
    }
#pragma unroll
    for (int off = 16; off >= 1; off >>= 1) mx = fmaxf(mx, __shfl_xor(mx, off, 32));
    float sum = 0.0f;
#pragma unroll
    for (int i = 0; i < 16; ++i) { vv[i] = __expf(vv[i] - mx); sum += vv[i]; }
#pragma unroll
    for (int off = 16; off >= 1; off >>= 1) sum += __shfl_xor(sum, off, 32);
    float inv = 1.0f / sum;
    size_t rbase = obase + (size_t)(strip * 16 + row) * SEQ;
#pragma unroll
    for (int i = 0; i < 16; ++i)
      out[rbase + lane + i * 32] = vv[i] * inv;
  }
}

extern "C" void kernel_launch(void* const* d_in, const int* in_sizes, int n_in,
                              void* d_out, int out_size, void* d_ws, size_t ws_size,
                              hipStream_t stream) {
  const float* x    = (const float*)d_in[0];
  const float* mask = (const float*)d_in[1];
  const float* Wq   = (const float*)d_in[2];
  const float* bq   = (const float*)d_in[3];
  const float* Wk   = (const float*)d_in[4];
  const float* bk   = (const float*)d_in[5];
  const float* Wv   = (const float*)d_in[6];
  const float* bv   = (const float*)d_in[7];
  (void)in_sizes; (void)n_in; (void)out_size; (void)ws_size;

  _Float16* ws16 = (_Float16*)d_ws;
  _Float16* xh   = ws16 + XH_OFF;
  _Float16* wt   = ws16 + WT_OFF;
  _Float16* qkv  = ws16 + QKV_OFF;
  float* out = (float*)d_out;

  cvt_x_kernel<<<(MTOT * HDIM) / 256, 256, 0, stream>>>(x, xh, MTOT * HDIM);
  wt_kernel<<<dim3(24, 24, 3), dim3(32, 8), 0, stream>>>(Wq, Wk, Wv, wt);
  proj_kernel<<<dim3(6, 32, 3), 256, 0, stream>>>(xh, wt, bq, bk, bv, qkv);
  attn_kernel<<<dim3(32, 12, 24), 256, 0, stream>>>(qkv, mask, out);
}
